// simple_MLPFit_53712861003849
// MI455X (gfx1250) — matrix operand order repaired and checked by reading, not compiled, not run on hardware
//
#include <hip/hip_runtime.h>
#include <hip/hip_fp16.h>

typedef __attribute__((ext_vector_type(16))) _Float16 v16h;
typedef __attribute__((ext_vector_type(8)))  _Float16 v8h;
typedef __attribute__((ext_vector_type(8)))  float    v8f;
typedef __attribute__((ext_vector_type(4)))  float    v4f;

#define WMMA_F16(A, B, C) \
    __builtin_amdgcn_wmma_f32_16x16x32_f16(false, (A), false, (B), (short)0, (C), false, false)

// Single-instruction ReLU (v_med3_f32): clamp(x, 0, +inf). Avoids the
// canonicalize+max pair that fmaxf() emits on WMMA outputs.
__device__ __forceinline__ float relu1(float v) {
    return __builtin_amdgcn_fmed3f(v, 0.0f, __builtin_inff());
}

// d_ws layout (bytes):
//   [0,    256)  : hist[64]   f32 accumulators
//   [256,  512)  : colsum[64] f32 accumulators
//   [1024, 9216) : W1 B-fragments, 8 frag x 32 lanes x 16 halves (f16)
//   [9216,17408) : F1 fragments
//   [17408,25600): F2 fragments
#define WS_FRAG_BYTE_OFF 1024
#define FRAG_HALVES_PER_MAT 4096   // 8*32*16

// ---------------------------------------------------------------------------
// Pack a 64x64 row-major weight matrix M (output h = a @ M^T, so B[k][n] = M[n][k])
// into per-lane WMMA B fragments: frag fi = t*2+kf (n0 = 16t, kb = 32kf).
// B layout (16-bit, 32x16): lane l, half i -> N = l%16, K = (i<8 ? i : i+8) + 8*(l/16).
// So half index i maps to K = 32*kf + 8*(l/16) + (i<8 ? i : i+8).
// ---------------------------------------------------------------------------
__global__ void k_prep(const float* __restrict__ W1,
                       const float* __restrict__ F1,
                       const float* __restrict__ F2,
                       float* __restrict__ ws)
{
    int tid = threadIdx.x;
    if (tid < 128) ws[tid] = 0.0f;   // zero hist + colsum accumulators

    _Float16* wh = (_Float16*)((char*)ws + WS_FRAG_BYTE_OFF);
    for (int g = tid; g < 3 * 8 * 32; g += blockDim.x) {
        int mat = g >> 8;            // 0..2
        int rem = g & 255;
        int fi  = rem >> 5;          // 0..7
        int l   = rem & 31;          // lane
        int t   = fi >> 1, kf = fi & 1;
        const float* M = (mat == 0) ? W1 : ((mat == 1) ? F1 : F2);
        int nrow  = 16 * t + (l & 15);
        int kbase = 32 * kf + 8 * (l >> 4);
        _Float16* dst = wh + mat * FRAG_HALVES_PER_MAT + fi * 512 + l * 16;
        #pragma unroll
        for (int i = 0; i < 16; ++i)
            dst[i] = (_Float16)M[nrow * 64 + kbase + (i < 8 ? i : i + 8)];
    }
}

// ---------------------------------------------------------------------------
// A-fragment layout (16-bit, 16x32): lane l, VGPR v, half p ->
//   M = l%16, K = 2*(v&3)+p + 16*(v>>2) + 8*(l>>4)
// => half i<8 : K = kb + i      + 8*(l>>4)
//    half i>=8: K = kb + i + 8  + 8*(l>>4)
// ---------------------------------------------------------------------------
__device__ __forceinline__ void build_layer0_frags(
    float xv, int hi, const float* sW, const float* sb, v16h A[2])
{
    #pragma unroll
    for (int kf = 0; kf < 2; ++kf) {
        int kb = kf * 32;
        #pragma unroll
        for (int i = 0; i < 16; ++i) {
            int j = kb + 8 * hi + (i < 8 ? i : i + 8);
            float h = fmaf(xv, sW[j], sb[j]);
            A[kf][i] = (_Float16)fmaxf(h, 0.0f);   // FMA result is canonical: 1 op
        }
    }
}

// ---------------------------------------------------------------------------
// Disc pass: weight1 = sigmoid(MLP_disc(x)); histogram(y, weight1) -> ws[0..63]
// Per 16-sample tile: batch all 8 WMMAs (4 independent chains), then relu-spill
// the h2 tile (f32) to per-wave LDS, re-read per-sample rows in contiguous 16B
// runs, dot with per-lane-reordered W2 in registers, one xor-16 shuffle,
// lanes 0-15 do sigmoid + hist atomic.
// ---------------------------------------------------------------------------
__global__ void __launch_bounds__(256) k_disc(
    const float* __restrict__ x, const float* __restrict__ y,
    const float* __restrict__ W0, const float* __restrict__ b0,
    const float* __restrict__ b1, const float* __restrict__ W2,
    const float* __restrict__ b2p, float* __restrict__ ws, int n)
{
    __shared__ float sW0[64], sb0[64], sb1[64], sW2[64], sHist[64];
    __shared__ __align__(16) float sT[8][16 * 72];   // per-wave h2 tile, padded rows
    int tid = threadIdx.x;
    if (tid < 64) {
        sW0[tid] = W0[tid]; sb0[tid] = b0[tid];
        sb1[tid] = b1[tid]; sW2[tid] = W2[tid];
        sHist[tid] = 0.0f;
    }
    __syncthreads();

    const _Float16* wh = (const _Float16*)((const char*)ws + WS_FRAG_BYTE_OFF);
    int lane = tid & 31, wave = tid >> 5;
    int m = lane & 15, hi = lane >> 4;
    float b2 = b2p[0];

    v16h B[8];
    #pragma unroll
    for (int fi = 0; fi < 8; ++fi)
        B[fi] = *(const v16h*)(wh + fi * 512 + lane * 16);

    float bias1[4];
    #pragma unroll
    for (int t = 0; t < 4; ++t) bias1[t] = sb1[16*t + m];

    // W2 reordered to this lane's A-fragment K order: q = kf*2 + r,
    // K = 32*kf + 16*r + 8*hi + e  (4 runs of 8)
    float w2k[4][8];
    #pragma unroll
    for (int q = 0; q < 4; ++q) {
        int kf = q >> 1, r = q & 1;
        int kbase = 32 * kf + 16 * r + 8 * hi;
        #pragma unroll
        for (int e = 0; e < 8; ++e) w2k[q][e] = sW2[kbase + e];
    }

    float* tP = sT[wave];
    int ntiles = n >> 4;
    for (int tile = blockIdx.x * 8 + wave; tile < ntiles; tile += gridDim.x * 8) {
        int base = tile * 16;
        float xv = x[base + m];

        v16h A[2];
        build_layer0_frags(xv, hi, sW0, sb0, A);

        // ---- phase 1: all 8 WMMAs, 4 independent accumulator chains
        v8f acc[4];
        #pragma unroll
        for (int t = 0; t < 4; ++t) {
            #pragma unroll
            for (int v = 0; v < 8; ++v) acc[t][v] = bias1[t];
            acc[t] = WMMA_F16(A[0], B[t*2 + 0], acc[t]);
            acc[t] = WMMA_F16(A[1], B[t*2 + 1], acc[t]);
        }

        // ---- phase 2: relu + spill to per-wave LDS tile
        #pragma unroll
        for (int t = 0; t < 4; ++t)
            #pragma unroll
            for (int v = 0; v < 8; ++v)
                tP[(v + 8*hi) * 72 + 16*t + m] = relu1(acc[t][v]);

        // ---- per-lane dot of sample-row m against W2 over this lane's K half
        float s = 0.0f;
        #pragma unroll
        for (int q = 0; q < 4; ++q) {
            int kf = q >> 1, r = q & 1;
            const float* p = tP + m * 72 + 32 * kf + 16 * r + 8 * hi;
            v4f a  = *(const v4f*)p;
            v4f b4 = *(const v4f*)(p + 4);
            #pragma unroll
            for (int e = 0; e < 4; ++e) s = fmaf(a[e],  w2k[q][e],     s);
            #pragma unroll
            for (int e = 0; e < 4; ++e) s = fmaf(b4[e], w2k[q][e + 4], s);
        }
        s += __shfl_xor(s, 16);      // merge the two half-wave K-partials

        if (hi == 0) {               // lanes 0-15: one sample each
            float w  = 1.0f / (1.0f + __expf(-(s + b2)));  // sigmoid
            float yv = y[base + m];
            if (yv >= 0.0f && yv < 1.0f) {
                int bn = (int)(yv * 64.0f);
                bn = bn > 63 ? 63 : bn;
                atomicAdd(&sHist[bn], w);
            }
        }
    }
    __syncthreads();
    if (tid < 64) atomicAdd(&ws[tid], sHist[tid]);
}

// ---------------------------------------------------------------------------
// Fit pass: accumulate column sums of g2 = relu(relu(relu(x@F0^T+f0)@F1^T+f1)@F2^T+f2)
// into ws[64..127]. (mean(x_fit) is linear, so F3/f3 are applied in k_final.)
// ---------------------------------------------------------------------------
__global__ void __launch_bounds__(256) k_fit(
    const float* __restrict__ x,
    const float* __restrict__ F0, const float* __restrict__ f0,
    const float* __restrict__ f1, const float* __restrict__ f2,
    float* __restrict__ ws, int n)
{
    __shared__ float sF0[64], sf0[64], sf1[64], sf2[64];
    __shared__ __align__(16) _Float16 sTile[8][16 * 80];  // per-wave C->A reshuffle
    int tid = threadIdx.x;
    if (tid < 64) { sF0[tid] = F0[tid]; sf0[tid] = f0[tid];
                    sf1[tid] = f1[tid]; sf2[tid] = f2[tid]; }
    __syncthreads();

    const _Float16* wh = (const _Float16*)((const char*)ws + WS_FRAG_BYTE_OFF);
    int lane = tid & 31, wave = tid >> 5;
    int m = lane & 15, hi = lane >> 4;

    v16h B1[8], B2[8];
    #pragma unroll
    for (int fi = 0; fi < 8; ++fi) {
        B1[fi] = *(const v16h*)(wh + 1*FRAG_HALVES_PER_MAT + fi * 512 + lane * 16);
        B2[fi] = *(const v16h*)(wh + 2*FRAG_HALVES_PER_MAT + fi * 512 + lane * 16);
    }
    float bias1[4], bias2[4];
    #pragma unroll
    for (int t = 0; t < 4; ++t) { bias1[t] = sf1[16*t + m]; bias2[t] = sf2[16*t + m]; }

    _Float16* tileP = sTile[wave];
    float colAcc[4] = {0.f, 0.f, 0.f, 0.f};

    int ntiles = n >> 4;
    for (int tile = blockIdx.x * 8 + wave; tile < ntiles; tile += gridDim.x * 8) {
        int base = tile * 16;
        float xv = x[base + m];

        v16h A[2];
        build_layer0_frags(xv, hi, sF0, sf0, A);

        // ---- layer F1: all 8 WMMAs first (4 independent chains)
        v8f acc[4];
        #pragma unroll
        for (int t = 0; t < 4; ++t) {
            #pragma unroll
            for (int v = 0; v < 8; ++v) acc[t][v] = bias1[t];
            acc[t] = WMMA_F16(A[0], B1[t*2 + 0], acc[t]);
            acc[t] = WMMA_F16(A[1], B1[t*2 + 1], acc[t]);
        }
        // relu + spill to per-wave LDS tile (C-layout rows)
        #pragma unroll
        for (int t = 0; t < 4; ++t)
            #pragma unroll
            for (int v = 0; v < 8; ++v)
                tileP[(v + 8*hi) * 80 + 16*t + m] = (_Float16)relu1(acc[t][v]);

        // ---- reshuffle LDS rows into A-fragment layout (two aligned 8-half runs)
        v16h A2[2];
        #pragma unroll
        for (int kf = 0; kf < 2; ++kf) {
            int kb = 32 * kf + 8 * hi;
            v8h lo  = *(const v8h*)(tileP + m * 80 + kb);
            v8h hiv = *(const v8h*)(tileP + m * 80 + kb + 16);
            #pragma unroll
            for (int i = 0; i < 8; ++i) { A2[kf][i] = lo[i]; A2[kf][i + 8] = hiv[i]; }
        }

        // ---- layer F2: all 8 WMMAs first, then relu + column-sum accumulation
        #pragma unroll
        for (int t = 0; t < 4; ++t) {
            #pragma unroll
            for (int v = 0; v < 8; ++v) acc[t][v] = bias2[t];
            acc[t] = WMMA_F16(A2[0], B2[t*2 + 0], acc[t]);
            acc[t] = WMMA_F16(A2[1], B2[t*2 + 1], acc[t]);
        }
        #pragma unroll
        for (int t = 0; t < 4; ++t) {
            float s = 0.0f;
            #pragma unroll
            for (int v = 0; v < 8; ++v) s += relu1(acc[t][v]);
            colAcc[t] += s;   // column n = 16t + m, summed over this tile's rows
        }
    }

    // merge the two half-waves (same column, different rows), then global atomics
    #pragma unroll
    for (int t = 0; t < 4; ++t) {
        float s = colAcc[t] + __shfl_xor(colAcc[t], 16);
        if (hi == 0) atomicAdd(&ws[64 + 16 * t + m], s);
    }
}

// ---------------------------------------------------------------------------
// Final: xpars = colsum @ F3^T / N + f3 ; chi2 over 64 bins
// ---------------------------------------------------------------------------
__global__ void k_final(const float* __restrict__ F3, const float* __restrict__ f3,
                        const float* __restrict__ ws, float* __restrict__ out, float invN)
{
    __shared__ float sCol[64], xp[4], red[64];
    int tid = threadIdx.x;   // 64 threads
    sCol[tid] = ws[64 + tid];
    __syncthreads();
    if (tid < 4) {
        float s = 0.0f;
        for (int n2 = 0; n2 < 64; ++n2) s += sCol[n2] * F3[tid * 64 + n2];
        xp[tid] = s * invN + f3[tid];
    }
    __syncthreads();
    const float DELTA = 1.0f / 64.0f;
    float yh    = ws[tid] * 64.0f;           // hist / DELTA
    float cc    = ((float)tid + 0.5f) * DELTA;
    float z     = (cc - xp[1]) / xp[2];
    float yval  = xp[0] * __expf(-0.5f * z * z) + xp[3];
    float yerr2 = yh * 64.0f;                // yhist / DELTA
    float d     = yh - yval;
    red[tid]    = d * d / (yerr2 + 1e-3f);
    __syncthreads();
    if (tid == 0) {
        float s = 0.0f;
        for (int i = 0; i < 64; ++i) s += red[i];
        out[0] = s;
    }
}

// ---------------------------------------------------------------------------
extern "C" void kernel_launch(void* const* d_in, const int* in_sizes, int n_in,
                              void* d_out, int out_size, void* d_ws, size_t ws_size,
                              hipStream_t stream)
{
    const float* x  = (const float*)d_in[0];
    const float* y  = (const float*)d_in[1];
    const float* W0 = (const float*)d_in[2];
    const float* b0 = (const float*)d_in[3];
    const float* W1 = (const float*)d_in[4];
    const float* b1 = (const float*)d_in[5];
    const float* W2 = (const float*)d_in[6];
    const float* b2 = (const float*)d_in[7];
    const float* F0 = (const float*)d_in[8];
    const float* f0 = (const float*)d_in[9];
    const float* F1 = (const float*)d_in[10];
    const float* f1 = (const float*)d_in[11];
    const float* F2 = (const float*)d_in[12];
    const float* f2 = (const float*)d_in[13];
    const float* F3 = (const float*)d_in[14];
    const float* f3 = (const float*)d_in[15];
    float* ws = (float*)d_ws;
    int n = in_sizes[0];

    hipLaunchKernelGGL(k_prep, dim3(1), dim3(256), 0, stream, W1, F1, F2, ws);
    const int blocks = 512;
    hipLaunchKernelGGL(k_disc, dim3(blocks), dim3(256), 0, stream,
                       x, y, W0, b0, b1, W2, b2, ws, n);
    hipLaunchKernelGGL(k_fit, dim3(blocks), dim3(256), 0, stream,
                       x, F0, f0, f1, f2, ws, n);
    hipLaunchKernelGGL(k_final, dim3(1), dim3(64), 0, stream,
                       F3, f3, ws, (float*)d_out, 1.0f / (float)n);
}
